// MoE_84378927497480
// MI455X (gfx1250) — matrix operand order repaired and checked by reading, not compiled, not run on hardware
//
#include <hip/hip_runtime.h>
#include <stdint.h>

// ---------------------------------------------------------------------------
// MoE top-2 gating + expert FFN for MI455X (gfx1250, wave32, WMMA bf16 + TDM)
// B=4, N=2048, D=1024, E=16, H=4096, C=256
// ---------------------------------------------------------------------------

#define MOE_B 4
#define MOE_N 2048
#define MOE_D 1024
#define MOE_E 16
#define MOE_H 4096
#define MOE_C 256
#define MOE_T (MOE_B * MOE_N)          // 8192 tokens
#define MOE_M (MOE_B * MOE_C)          // 1024 rows per expert

typedef __attribute__((ext_vector_type(16))) __bf16 v16bf;
typedef __attribute__((ext_vector_type(8)))  __bf16 v8bf;
typedef __attribute__((ext_vector_type(8)))  float  v8f;
typedef __attribute__((ext_vector_type(4)))  unsigned int u32x4;
typedef __attribute__((ext_vector_type(8)))  unsigned int u32x8;

// native bf16 conversion (RNE) -> lets the backend use v_cvt_*_bf16_f32
__device__ __forceinline__ unsigned short f2bf(float x) {
    return __builtin_bit_cast(unsigned short, (__bf16)x);
}

__device__ __forceinline__ v16bf load_frag2(const unsigned short* p0,
                                            const unsigned short* p1) {
    v8bf lo = *(const v8bf*)p0;
    v8bf hi = *(const v8bf*)p1;
    v16bf r;
#pragma unroll
    for (int i = 0; i < 8; ++i) { r[i] = lo[i]; r[i + 8] = hi[i]; }
    return r;
}

// ---------------------------------------------------------------------------
// TDM: DMA a 128-row x 32-elem bf16 tile (row stride lda_elems) from global
// into LDS, padding each 64B row with 16B so LDS rows land at 40-elem stride.
// D# per CDNA5 ISA; 2D tensor -> 2 SGPR groups.
// ---------------------------------------------------------------------------
__device__ __forceinline__ void tdm_load_a_tile(const unsigned short* gsrc,
                                                unsigned short* lds_dst,
                                                unsigned lda_elems) {
    unsigned long long ga = (unsigned long long)(uintptr_t)gsrc;
    unsigned ga_lo = __builtin_amdgcn_readfirstlane((unsigned)ga);
    unsigned ga_hi = __builtin_amdgcn_readfirstlane((unsigned)(ga >> 32) & 0x1FFFFFFu);
    unsigned la    = __builtin_amdgcn_readfirstlane((unsigned)(uintptr_t)lds_dst);
    unsigned ld    = __builtin_amdgcn_readfirstlane(lda_elems);

    u32x4 g0;
    g0[0] = 1u;                        // count=1 (valid user descriptor)
    g0[1] = la;                        // lds_addr (bytes)
    g0[2] = ga_lo;                     // global_addr[31:0]
    g0[3] = ga_hi | (2u << 30);        // global_addr[56:32] | type=2 ("image")

    u32x8 g1;
    // data_size=1 (2B) | pad_enable | pad_interval=3 (16 DWORDs) | pad_amount=3 (4 DWORDs)
    g1[0] = (1u << 16) | (1u << 20) | (3u << 22) | (3u << 25);
    g1[1] = 0u;                        // atomic_barrier_addr=0 | tensor_dim0[15:0] (dim0=1<<30)
    g1[2] = 0x4000u;                   // tensor_dim0[31:16]    | tensor_dim1[15:0] (dim1=1<<30)
    g1[3] = 0x4000u | (32u << 16);     // tensor_dim1[31:16]    | tile_dim0=32
    g1[4] = 128u;                      // tile_dim1=128         | tile_dim2=0
    g1[5] = ld;                        // tensor_dim0_stride[31:0]
    g1[6] = 0u;                        // stride0[47:32] | stride1[15:0]
    g1[7] = 0u;                        // stride1[47:16]

    asm volatile("tensor_load_to_lds %0, %1" :: "s"(g0), "s"(g1) : "memory");
}

// ---------------------------------------------------------------------------
// generic u32 fill
// ---------------------------------------------------------------------------
__global__ void moe_fill_u32(unsigned int* __restrict__ p, unsigned int v, size_t n) {
    size_t i = (size_t)blockIdx.x * blockDim.x + threadIdx.x;
    size_t s = (size_t)gridDim.x * blockDim.x;
    for (; i < n; i += s) p[i] = v;
}

// ---------------------------------------------------------------------------
// gating: one wave per token. logits = x @ wg, softmax, top-2, noise policy.
// ---------------------------------------------------------------------------
__global__ __launch_bounds__(256) void moe_gating(
    const float* __restrict__ x, const float* __restrict__ wg,
    const float* __restrict__ noise,
    float* __restrict__ raw_gates, int* __restrict__ idx1, int* __restrict__ idx2,
    float* __restrict__ g1n, float* __restrict__ g2n, int* __restrict__ keep2) {
    const int lane = threadIdx.x & 31;
    const int wid  = threadIdx.x >> 5;
    const int t    = blockIdx.x * 8 + wid;             // token id
    const float* xr = x + (size_t)t * MOE_D;

    float acc[MOE_E];
#pragma unroll
    for (int e = 0; e < MOE_E; ++e) acc[e] = 0.f;

    for (int d0 = 0; d0 < MOE_D; d0 += 32) {
        float xv = xr[d0 + lane];
        const float* wrow = wg + (size_t)(d0 + lane) * MOE_E;
#pragma unroll
        for (int e = 0; e < MOE_E; ++e) acc[e] += xv * wrow[e];
    }
#pragma unroll
    for (int e = 0; e < MOE_E; ++e) {
        float v = acc[e];
        v += __shfl_xor(v, 16, 32);
        v += __shfl_xor(v, 8, 32);
        v += __shfl_xor(v, 4, 32);
        v += __shfl_xor(v, 2, 32);
        v += __shfl_xor(v, 1, 32);
        acc[e] = v;
    }
    if (lane == 0) {
        float mx = acc[0];
#pragma unroll
        for (int e = 1; e < MOE_E; ++e) mx = fmaxf(mx, acc[e]);
        float p[MOE_E], se = 0.f;
#pragma unroll
        for (int e = 0; e < MOE_E; ++e) { p[e] = __expf(acc[e] - mx); se += p[e]; }
        float inv = 1.f / se;
#pragma unroll
        for (int e = 0; e < MOE_E; ++e) {
            p[e] *= inv;
            raw_gates[(size_t)t * MOE_E + e] = p[e];
        }
        int i1 = 0; float g1v = p[0];
#pragma unroll
        for (int e = 1; e < MOE_E; ++e)
            if (p[e] > g1v) { g1v = p[e]; i1 = e; }
        int i2 = 0; float g2v = (0 == i1) ? 0.f : p[0];
#pragma unroll
        for (int e = 1; e < MOE_E; ++e) {
            float v = (e == i1) ? 0.f : p[e];
            if (v > g2v) { g2v = v; i2 = e; }
        }
        float denom = g1v + g2v + 1e-9f;
        float g1 = g1v / denom, g2 = g2v / denom;
        idx1[t] = i1; idx2[t] = i2;
        g1n[t] = g1;  g2n[t] = g2;
        keep2[t] = (noise[t] < (g2 / 0.2f)) ? 1 : 0;
    }
}

// ---------------------------------------------------------------------------
// capacity scan: one wave per (b,e). Ballot-based exclusive prefix over N.
// ---------------------------------------------------------------------------
__global__ __launch_bounds__(32) void moe_scan(
    const int* __restrict__ idx1, const int* __restrict__ idx2,
    const int* __restrict__ keep2, int* __restrict__ dest1, int* __restrict__ dest2) {
    const int lane = threadIdx.x;
    const int b = blockIdx.x >> 4;
    const int e = blockIdx.x & 15;
    const int base = b * MOE_N;
    const int slot0 = e * MOE_M + b * MOE_C;

    int run = 0;
    for (int c = 0; c < MOE_N; c += 32) {
        int t = base + c + lane;
        bool f = (idx1[t] == e);
        unsigned m = (unsigned)__ballot(f);
        int pre = __popc(m & ((1u << lane) - 1u));
        if (f) {
            int pos = run + pre;
            dest1[t] = (pos < MOE_C) ? (slot0 + pos) : -1;
        }
        run += __popc(m);
    }
    int count1 = run < MOE_C ? run : MOE_C;            // mask_1_count

    int run2 = count1;
    for (int c = 0; c < MOE_N; c += 32) {
        int t = base + c + lane;
        bool own = (idx2[t] == e);
        bool f = own && (keep2[t] != 0);
        unsigned m = (unsigned)__ballot(f);
        int pre = __popc(m & ((1u << lane) - 1u));
        if (own) {
            int pos = run2 + pre;
            dest2[t] = (f && pos < MOE_C) ? (slot0 + pos) : -1;
        }
        run2 += __popc(m);
    }
}

// ---------------------------------------------------------------------------
// dispatch: gather token rows (fp32 -> bf16) into expert slots; inverse map
// ---------------------------------------------------------------------------
__global__ __launch_bounds__(256) void moe_dispatch(
    const float* __restrict__ x, const int* __restrict__ dest1,
    const int* __restrict__ dest2, const float* __restrict__ g1n,
    const float* __restrict__ g2n, unsigned short* __restrict__ Xe,
    int* __restrict__ tok_of_slot, float* __restrict__ gate_of_slot) {
    const int t = blockIdx.x;
    const float* xr = x + (size_t)t * MOE_D;

    int s = dest1[t]; float g = g1n[t];
    if (s >= 0 && g != 0.f) {
        unsigned short* dr = Xe + (size_t)s * MOE_D;
        for (int i = threadIdx.x; i < MOE_D; i += 256) dr[i] = f2bf(xr[i]);
        if (threadIdx.x == 0) { tok_of_slot[s] = t; gate_of_slot[s] = g; }
    }
    s = dest2[t]; g = g2n[t];
    if (s >= 0 && g != 0.f) {
        unsigned short* dr = Xe + (size_t)s * MOE_D;
        for (int i = threadIdx.x; i < MOE_D; i += 256) dr[i] = f2bf(xr[i]);
        if (threadIdx.x == 0) { tok_of_slot[s] = t; gate_of_slot[s] = g; }
    }
}

// ---------------------------------------------------------------------------
// shared double-buffered bf16 GEMM core: 128x256 block tile, 8 waves,
// wave tile 64x64 = 4x4 WMMA 16x16x32 bf16 frags. A tile via TDM DMA,
// B tile via fp32->bf16 register staging (transpose into sB[n][k]).
// ---------------------------------------------------------------------------
template<int K, int LDA, int LDB>
__device__ __forceinline__ void gemm_core(const unsigned short* __restrict__ A,
                                          const float* __restrict__ Bw,
                                          int m0, int n0, v8f (&acc)[4][4]) {
    __shared__ unsigned short sA[2][128 * 40];
    __shared__ unsigned short sB[2][256 * 40];

    const int tid  = threadIdx.x;
    const int lane = tid & 31;
    const int wid  = tid >> 5;
    const int wr   = (wid & 1) * 64;
    const int wc   = (wid >> 1) * 64;
    const int lm   = lane & 15;
    const int half = lane >> 4;
    const int kr   = tid >> 3;          // B-tile: K row 0..31
    const int nc0  = (tid & 7) * 32;    // B-tile: 32 consecutive N per thread

    // ---- prologue: tile 0 ----
    if (wid == 0) tdm_load_a_tile(A + (size_t)m0 * LDA, sA[0], LDA);
    {
        const float* src = Bw + (size_t)kr * LDB + n0 + nc0;
#pragma unroll
        for (int i = 0; i < 32; i += 4) {
            float4 f = *(const float4*)(src + i);
            sB[0][(nc0 + i + 0) * 40 + kr] = f2bf(f.x);
            sB[0][(nc0 + i + 1) * 40 + kr] = f2bf(f.y);
            sB[0][(nc0 + i + 2) * 40 + kr] = f2bf(f.z);
            sB[0][(nc0 + i + 3) * 40 + kr] = f2bf(f.w);
        }
    }
    if (wid == 0) __builtin_amdgcn_s_wait_tensorcnt(0);
    __syncthreads();

    const int KT = K / 32;
    for (int kt = 0; kt < KT; ++kt) {
        const int cur = kt & 1, nxt = cur ^ 1;
        const bool more = (kt + 1 < KT);
        const int k1 = (kt + 1) * 32;

        // ---- issue next tile's loads (TDM for A, registers for B) ----
        float4 f[8];
        if (more) {
            if (wid == 0)
                tdm_load_a_tile(A + (size_t)m0 * LDA + k1, sA[nxt], LDA);
            const float* src = Bw + (size_t)(k1 + kr) * LDB + n0 + nc0;
#pragma unroll
            for (int i = 0; i < 8; ++i) f[i] = *(const float4*)(src + 4 * i);
        }

        // ---- compute on current tile ----
        v16bf af[4], bf[4];
#pragma unroll
        for (int i = 0; i < 4; ++i) {
            const unsigned short* base = sA[cur] + (wr + i * 16 + lm) * 40 + half * 8;
            af[i] = load_frag2(base, base + 16);      // K 0-7/16-23 vs 8-15/24-31
        }
#pragma unroll
        for (int j = 0; j < 4; ++j) {
            const unsigned short* base = sB[cur] + (wc + j * 16 + lm) * 40 + half * 8;
            bf[j] = load_frag2(base, base + 16);      // K 0-7/16-23 vs 8-15/24-31
        }
#pragma unroll
        for (int i = 0; i < 4; ++i)
#pragma unroll
            for (int j = 0; j < 4; ++j)
                acc[i][j] = __builtin_amdgcn_wmma_f32_16x16x32_bf16(
                    false, af[i], false, bf[j], (short)0, acc[i][j], false, false);

        // ---- drain next tile into LDS, wait DMA, flip ----
        if (more) {
#pragma unroll
            for (int i = 0; i < 8; ++i) {
                sB[nxt][(nc0 + 4 * i + 0) * 40 + kr] = f2bf(f[i].x);
                sB[nxt][(nc0 + 4 * i + 1) * 40 + kr] = f2bf(f[i].y);
                sB[nxt][(nc0 + 4 * i + 2) * 40 + kr] = f2bf(f[i].z);
                sB[nxt][(nc0 + 4 * i + 3) * 40 + kr] = f2bf(f[i].w);
            }
            if (wid == 0) __builtin_amdgcn_s_wait_tensorcnt(0);
        }
        __syncthreads();
    }
}

// ---------------------------------------------------------------------------
// GEMM1: hidden[e] = relu(Xe[e](1024x1024 bf16) @ w1[e](1024x4096 f32->bf16))
// ---------------------------------------------------------------------------
__global__ __launch_bounds__(256) void moe_gemm1(
    const unsigned short* __restrict__ Xe, const float* __restrict__ w1,
    unsigned short* __restrict__ hidden) {
    const int e  = blockIdx.z;
    const int m0 = blockIdx.x * 128;
    const int n0 = blockIdx.y * 256;
    const unsigned short* A = Xe + (size_t)e * MOE_M * MOE_D;
    const float* Bw = w1 + (size_t)e * MOE_D * MOE_H;
    unsigned short* Ho = hidden + (size_t)e * MOE_M * MOE_H;

    v8f acc[4][4];
    const v8f vz = {0.f, 0.f, 0.f, 0.f, 0.f, 0.f, 0.f, 0.f};
#pragma unroll
    for (int i = 0; i < 4; ++i)
#pragma unroll
        for (int j = 0; j < 4; ++j) acc[i][j] = vz;

    gemm_core<MOE_D, MOE_D, MOE_H>(A, Bw, m0, n0, acc);

    const int lane = threadIdx.x & 31, wid = threadIdx.x >> 5;
    const int wr = (wid & 1) * 64, wc = (wid >> 1) * 64;
    const int lm = lane & 15, rbase = (lane >> 4) * 8;
#pragma unroll
    for (int i = 0; i < 4; ++i)
#pragma unroll
        for (int j = 0; j < 4; ++j) {
            int col = n0 + wc + j * 16 + lm;
#pragma unroll
            for (int r = 0; r < 8; ++r) {
                int row = m0 + wr + i * 16 + rbase + r;
                float v = acc[i][j][r];
                Ho[(size_t)row * MOE_H + col] = f2bf(v > 0.f ? v : 0.f);
            }
        }
}

// ---------------------------------------------------------------------------
// GEMM2: y[e] = hidden[e](1024x4096 bf16) @ w2[e](4096x1024 f32->bf16)
// epilogue: out[token] += gate * y (atomic f32; <=2 commutative adds/elem)
// ---------------------------------------------------------------------------
__global__ __launch_bounds__(256) void moe_gemm2(
    const unsigned short* __restrict__ hidden, const float* __restrict__ w2,
    const int* __restrict__ tok_of_slot, const float* __restrict__ gate_of_slot,
    float* __restrict__ out) {
    const int e  = blockIdx.z;
    const int m0 = blockIdx.x * 128;
    const int n0 = blockIdx.y * 256;
    const unsigned short* A = hidden + (size_t)e * MOE_M * MOE_H;
    const float* Bw = w2 + (size_t)e * MOE_H * MOE_D;

    v8f acc[4][4];
    const v8f vz = {0.f, 0.f, 0.f, 0.f, 0.f, 0.f, 0.f, 0.f};
#pragma unroll
    for (int i = 0; i < 4; ++i)
#pragma unroll
        for (int j = 0; j < 4; ++j) acc[i][j] = vz;

    gemm_core<MOE_H, MOE_H, MOE_D>(A, Bw, m0, n0, acc);

    const int lane = threadIdx.x & 31, wid = threadIdx.x >> 5;
    const int wr = (wid & 1) * 64, wc = (wid >> 1) * 64;
    const int lm = lane & 15, rbase = (lane >> 4) * 8;
#pragma unroll
    for (int i = 0; i < 4; ++i) {
        int tokr[8]; float gr[8];
#pragma unroll
        for (int r = 0; r < 8; ++r) {
            int sidx = e * MOE_M + m0 + wr + i * 16 + rbase + r;
            tokr[r] = tok_of_slot[sidx];
            gr[r]   = gate_of_slot[sidx];
        }
#pragma unroll
        for (int j = 0; j < 4; ++j) {
            int col = n0 + wc + j * 16 + lm;
#pragma unroll
            for (int r = 0; r < 8; ++r) {
                if (tokr[r] >= 0) {
                    atomicAdd(out + (size_t)tokr[r] * MOE_D + col, gr[r] * acc[i][j][r]);
                }
            }
        }
    }
}

// ---------------------------------------------------------------------------
// loss: mean_{b,e}(proxy_mean * density) * E^2 * 0.01, fixed-order reduction
// ---------------------------------------------------------------------------
__global__ __launch_bounds__(64) void moe_loss(
    const float* __restrict__ raw_gates, const int* __restrict__ idx1,
    float* __restrict__ out_loss) {
    __shared__ float part[64];
    const int tid = threadIdx.x;            // 64 threads: (b,e) pairs
    const int b = tid >> 4, e = tid & 15;
    float proxy = 0.f; int cnt = 0;
    for (int n = 0; n < MOE_N; ++n) {
        int t = b * MOE_N + n;
        proxy += raw_gates[(size_t)t * MOE_E + e];
        cnt += (idx1[t] == e) ? 1 : 0;
    }
    part[tid] = (proxy / (float)MOE_N) * ((float)cnt / (float)MOE_N);
    __syncthreads();
    if (tid == 0) {
        float s = 0.f;
        for (int i = 0; i < 64; ++i) s += part[i];
        *out_loss = (s / 64.f) * (float)(MOE_E * MOE_E) * 0.01f;
    }
}

// ---------------------------------------------------------------------------
extern "C" void kernel_launch(void* const* d_in, const int* in_sizes, int n_in,
                              void* d_out, int out_size, void* d_ws, size_t ws_size,
                              hipStream_t stream) {
    (void)in_sizes; (void)n_in; (void)ws_size;
    const float* x     = (const float*)d_in[0];
    const float* wg    = (const float*)d_in[1];
    const float* w1    = (const float*)d_in[2];
    const float* w2    = (const float*)d_in[3];
    const float* noise = (const float*)d_in[4];
    float* out = (float*)d_out;                     // [B*N*D] floats + 1 loss float

    uint8_t* wsb = (uint8_t*)d_ws;
    size_t off = 0;
    auto take = [&](size_t nbytes) -> void* {
        void* p = wsb + off;
        off += (nbytes + 255) & ~(size_t)255;
        return p;
    };
    float* raw_gates = (float*)take((size_t)MOE_T * MOE_E * 4);
    int*   idx1  = (int*)take((size_t)MOE_T * 4);
    int*   idx2  = (int*)take((size_t)MOE_T * 4);
    float* g1n   = (float*)take((size_t)MOE_T * 4);
    float* g2n   = (float*)take((size_t)MOE_T * 4);
    int*   keep2 = (int*)take((size_t)MOE_T * 4);
    int*   dest1 = (int*)take((size_t)MOE_T * 4);
    int*   dest2 = (int*)take((size_t)MOE_T * 4);
    int*   tok_of_slot  = (int*)take((size_t)MOE_E * MOE_M * 4);
    float* gate_of_slot = (float*)take((size_t)MOE_E * MOE_M * 4);
    unsigned short* Xe     = (unsigned short*)take((size_t)MOE_E * MOE_M * MOE_D * 2);
    unsigned short* hidden = (unsigned short*)take((size_t)MOE_E * MOE_M * MOE_H * 2);

    // init: Xe=0, tok=-1, gate=0, out=0 (incl. loss slot)
    moe_fill_u32<<<4096, 256, 0, stream>>>((unsigned int*)Xe, 0u,
                                           (size_t)MOE_E * MOE_M * MOE_D / 2);
    moe_fill_u32<<<64, 256, 0, stream>>>((unsigned int*)tok_of_slot, 0xFFFFFFFFu,
                                         (size_t)MOE_E * MOE_M);
    moe_fill_u32<<<64, 256, 0, stream>>>((unsigned int*)gate_of_slot, 0u,
                                         (size_t)MOE_E * MOE_M);
    moe_fill_u32<<<4096, 256, 0, stream>>>((unsigned int*)out, 0u, (size_t)out_size);

    moe_gating<<<MOE_T / 8, 256, 0, stream>>>(x, wg, noise, raw_gates,
                                              idx1, idx2, g1n, g2n, keep2);
    moe_scan<<<MOE_B * MOE_E, 32, 0, stream>>>(idx1, idx2, keep2, dest1, dest2);
    moe_dispatch<<<MOE_T, 256, 0, stream>>>(x, dest1, dest2, g1n, g2n,
                                            Xe, tok_of_slot, gate_of_slot);
    moe_gemm1<<<dim3(MOE_M / 128, MOE_H / 256, MOE_E), 256, 0, stream>>>(Xe, w1, hidden);
    moe_gemm2<<<dim3(MOE_M / 128, MOE_D / 256, MOE_E), 256, 0, stream>>>(
        hidden, w2, tok_of_slot, gate_of_slot, out);
    moe_loss<<<1, 64, 0, stream>>>(raw_gates, idx1, out + (size_t)MOE_B * MOE_N * MOE_D);
}
